// MATTN_38955353375091
// MI455X (gfx1250) — matrix operand order repaired and checked by reading, not compiled, not run on hardware
//
#include <hip/hip_runtime.h>
#include <hip/hip_bf16.h>

#define NB 32
#define CCH 256
#define HH 48
#define HW 2304      // 48*48
#define TT 110
#define TP 128       // padded token count
#define OHW 46
#define OP 2116      // 46*46
#define KCONV 4608   // 512*9

#ifndef __has_builtin
#define __has_builtin(x) 0
#endif
#if __has_builtin(__builtin_amdgcn_global_load_async_to_lds_b128) && \
    __has_builtin(__builtin_amdgcn_s_wait_asynccnt)
#define USE_ASYNC_LDS 1
#else
#define USE_ASYNC_LDS 0
#endif

typedef __attribute__((ext_vector_type(16))) __bf16 v16bf;
typedef __attribute__((ext_vector_type(8)))  float  v8f;
typedef __attribute__((ext_vector_type(4)))  int    v4i;

__device__ __forceinline__ unsigned short f2bf(float f) {
    union { float f; unsigned u; } v; v.f = f;
    unsigned r = v.u + 0x7FFFu + ((v.u >> 16) & 1u);
    return (unsigned short)(r >> 16);
}
__device__ __forceinline__ __bf16 bfbits(unsigned short h) {
    union { unsigned short u; __bf16 b; } cv; cv.u = h; return cv.b;
}

#if USE_ASYNC_LDS
// async copy of one 16B chunk global -> LDS (tracked by ASYNCcnt)
__device__ __forceinline__ void async_cp16(const unsigned short* g, unsigned short* l) {
    __builtin_amdgcn_global_load_async_to_lds_b128(
        (__attribute__((address_space(1))) v4i*)(g),
        (__attribute__((address_space(3))) v4i*)(l), 0, 0);
}
#endif

// ---- WMMA fragment gathers per CDNA5 ISA 7.12.2 (wave32) ----
// A tile in LDS: row-major [rows][32] ushort(bf16). 16x32 fragment at row base mrow.
__device__ __forceinline__ v16bf frag_a(const unsigned short* As, int mrow, int lane) {
    const unsigned short* rp = As + (mrow + (lane & 15)) * 32;
    int kh = (lane >> 4) * 8;
    v16bf a;
#pragma unroll
    for (int e = 0; e < 16; ++e) {
        int v = e >> 1;
        int k = ((v & 3) * 2) + ((v >> 2) * 16) + kh + (e & 1);
        a[e] = bfbits(rp[k]);
    }
    return a;
}
// B tile in LDS: N-major [64 cols][32 k] ushort. 32x16 fragment at col base ncol.
__device__ __forceinline__ v16bf frag_b(const unsigned short* Bs, int ncol, int lane) {
    const unsigned short* cp = Bs + (ncol + (lane & 15)) * 32;
    int kh = (lane >> 4) * 8;
    v16bf b;
#pragma unroll
    for (int e = 0; e < 16; ++e) b[e] = bfbits(cp[e + kh + ((e >> 3) << 3)]);
    return b;
}

// ---- pyramid pooling: x [n][c][48][48] f32 -> pooled [n][c][128] f32 (t>=110 zero)
__global__ __launch_bounds__(128)
void k_pool(const float* __restrict__ x, float* __restrict__ pooled) {
    int c = blockIdx.x, n = blockIdx.y;
    __shared__ float tile[HW];
    const float* g = x + ((long long)n * CCH + c) * HW;
    for (int i = threadIdx.x; i < HW; i += 128) tile[i] = g[i];
    __syncthreads();
    int t = threadIdx.x;
    float r = 0.f;
    if (t < TT) {
        int o, j;
        if (t == 0)      { o = 1; j = 0; }
        else if (t < 10) { o = 3; j = t - 1; }
        else if (t < 46) { o = 6; j = t - 10; }
        else             { o = 8; j = t - 46; }
        int k = HH / o;
        int iy = (j / o) * k, ix = (j % o) * k;
        float s = 0.f;
        for (int yy = 0; yy < k; ++yy)
            for (int xx = 0; xx < k; ++xx)
                s += tile[(iy + yy) * HH + ix + xx];
        r = s / (float)(k * k);
    }
    pooled[((long long)n * CCH + c) * TP + t] = r;
}

// ---- small projections on pooled tensors + relu + token-L2-norm -> bf16 [n][128][C]
__global__ __launch_bounds__(128)
void k_sproj(const float* __restrict__ Wm1, const float* __restrict__ Wm3,
             const float* __restrict__ Wl1, const float* __restrict__ Wl3,
             const float* __restrict__ p1,  const float* __restrict__ p2,
             unsigned short* SAM, unsigned short* SM3,
             unsigned short* SAL, unsigned short* SL3) {
    int c = blockIdx.x, pair = blockIdx.y, n = blockIdx.z;
    const float* W = (pair == 0) ? Wm1 : (pair == 1) ? Wm3 : (pair == 2) ? Wl1 : Wl3;
    const float* P = (pair < 2) ? p1 : p2;
    unsigned short* O = (pair == 0) ? SAM : (pair == 1) ? SM3 : (pair == 2) ? SAL : SL3;
    int t = threadIdx.x;
    const float* Wr = W + c * CCH;
    const float* Pn = P + (long long)n * CCH * TP;
    float s = 0.f;
    for (int k = 0; k < CCH; ++k) s += Wr[k] * Pn[k * TP + t];
    s = s > 0.f ? s : 0.f;
    __shared__ float red[128];
    red[t] = s * s;
    __syncthreads();
    for (int st = 64; st > 0; st >>= 1) { if (t < st) red[t] += red[t + st]; __syncthreads(); }
    float inv = 1.f / fmaxf(sqrtf(red[0]), 1e-12f);
    O[((long long)n * TP + t) * CCH + c] = f2bf(s * inv);
}

// ---- generic bf16 WMMA GEMM: C[z] = A[z](MxK) * B[z](KxN) (+ A2*B2 when dual)
// strides are in elements. A/B may be f32 (converted on stage) or bf16.
__global__ __launch_bounds__(256)
void k_gemm(const void* Ap, long long sA, int lda, int aF32,
            const void* Bp, long long sB, int ldb, int bF32,
            const void* A2p, long long sA2,
            const void* B2p, long long sB2, int dual,
            void* Cp, long long sC, int ldc, int outBf16,
            int K) {
    __shared__ __align__(16) unsigned short As[128 * 32];
    __shared__ __align__(16) unsigned short Bs[64 * 32];
    __shared__ __align__(16) unsigned short As2[128 * 32];
    __shared__ __align__(16) unsigned short Bs2[64 * 32];
    int m0 = blockIdx.x * 128, n0 = blockIdx.y * 64, z = blockIdx.z;
    int tid = threadIdx.x, lane = tid & 31, w = tid >> 5;
    int wm = (w & 1) * 64, wn = (w >> 1) * 16;
    v8f acc[4];
    for (int i = 0; i < 4; ++i) acc[i] = (v8f){0.f,0.f,0.f,0.f,0.f,0.f,0.f,0.f};
    for (int k0 = 0; k0 < K; k0 += 32) {
        __syncthreads();
        {   // stage A: 128x32, contiguous rows
            int row = tid >> 1, kb = (tid & 1) * 16;
            long long base = (long long)(m0 + row) * lda + k0 + kb;
            if (aF32) {
                const float* g = (const float*)Ap + sA * z + base;
#pragma unroll
                for (int j = 0; j < 16; ++j) As[row * 32 + kb + j] = f2bf(g[j]);
                if (k0 + 32 < K) __builtin_prefetch((const void*)(g + 32), 0, 0);
            } else {
#if USE_ASYNC_LDS
                const unsigned short* g = (const unsigned short*)Ap + sA * z + base;
                async_cp16(g, &As[row * 32 + kb]);
                async_cp16(g + 8, &As[row * 32 + kb + 8]);
#else
                const unsigned short* g = (const unsigned short*)Ap + sA * z + base;
#pragma unroll
                for (int j = 0; j < 16; ++j) As[row * 32 + kb + j] = g[j];
#endif
            }
            if (dual) {
                const unsigned short* g = (const unsigned short*)A2p + sA2 * z + base;
#if USE_ASYNC_LDS
                async_cp16(g, &As2[row * 32 + kb]);
                async_cp16(g + 8, &As2[row * 32 + kb + 8]);
#else
#pragma unroll
                for (int j = 0; j < 16; ++j) As2[row * 32 + kb + j] = g[j];
#endif
            }
        }
        {   // stage B: 32x64 -> LDS N-major [col][k]; per-thread contiguous-k chunk,
            // per-j coalesced global reads, single b128 LDS store.
            int col = tid & 63;
            int kb = (tid >> 6) * 8;
            long long base = (long long)(k0 + kb) * ldb + n0 + col;
            unsigned short tmp[8];
            if (bF32) {
                const float* g = (const float*)Bp + sB * z + base;
#pragma unroll
                for (int j = 0; j < 8; ++j) tmp[j] = f2bf(g[(long long)j * ldb]);
                if (k0 + 32 < K) __builtin_prefetch((const void*)(g + 32LL * ldb), 0, 0);
            } else {
                const unsigned short* g = (const unsigned short*)Bp + sB * z + base;
#pragma unroll
                for (int j = 0; j < 8; ++j) tmp[j] = g[(long long)j * ldb];
            }
            uint4 vv;
            vv.x = (unsigned)tmp[0] | ((unsigned)tmp[1] << 16);
            vv.y = (unsigned)tmp[2] | ((unsigned)tmp[3] << 16);
            vv.z = (unsigned)tmp[4] | ((unsigned)tmp[5] << 16);
            vv.w = (unsigned)tmp[6] | ((unsigned)tmp[7] << 16);
            *(uint4*)(&Bs[col * 32 + kb]) = vv;
            if (dual) {
                const unsigned short* g = (const unsigned short*)B2p + sB2 * z + base;
#pragma unroll
                for (int j = 0; j < 8; ++j) tmp[j] = g[(long long)j * ldb];
                vv.x = (unsigned)tmp[0] | ((unsigned)tmp[1] << 16);
                vv.y = (unsigned)tmp[2] | ((unsigned)tmp[3] << 16);
                vv.z = (unsigned)tmp[4] | ((unsigned)tmp[5] << 16);
                vv.w = (unsigned)tmp[6] | ((unsigned)tmp[7] << 16);
                *(uint4*)(&Bs2[col * 32 + kb]) = vv;
            }
        }
#if USE_ASYNC_LDS
        if (!aF32) __builtin_amdgcn_s_wait_asynccnt(0);
#endif
        __syncthreads();
        v16bf b = frag_b(Bs, wn, lane);
#pragma unroll
        for (int mi = 0; mi < 4; ++mi) {
            v16bf a = frag_a(As, wm + mi * 16, lane);
            acc[mi] = __builtin_amdgcn_wmma_f32_16x16x32_bf16(
                false, a, false, b, (short)0, acc[mi], false, false);
        }
        if (dual) {
            v16bf b2 = frag_b(Bs2, wn, lane);
#pragma unroll
            for (int mi = 0; mi < 4; ++mi) {
                v16bf a2 = frag_a(As2, wm + mi * 16, lane);
                acc[mi] = __builtin_amdgcn_wmma_f32_16x16x32_bf16(
                    false, a2, false, b2, (short)0, acc[mi], false, false);
            }
        }
    }
    int col = n0 + wn + (lane & 15);
    int kh = (lane >> 4) * 8;
#pragma unroll
    for (int mi = 0; mi < 4; ++mi)
#pragma unroll
        for (int i = 0; i < 8; ++i) {
            int row = m0 + wm + mi * 16 + i + kh;
            if (outBf16)
                ((unsigned short*)Cp + sC * z)[(long long)row * ldc + col] = f2bf(acc[mi][i]);
            else
                ((float*)Cp + sC * z)[(long long)row * ldc + col] = acc[mi][i];
        }
}

// ---- softmax over 110 tokens: logits [n][128][hw] f32 -> Abf [n][hw][128] bf16 (padded 0)
__global__ __launch_bounds__(128)
void k_softmax(const float* __restrict__ logits, unsigned short* __restrict__ Abf) {
    int n = blockIdx.y;
    int s = blockIdx.x * 128 + threadIdx.x;
    const float* L = logits + (long long)n * TP * HW + s;
    float mx = -1e30f;
    for (int t = 0; t < TT; ++t) mx = fmaxf(mx, L[(long long)t * HW]);
    float sum = 0.f;
    for (int t = 0; t < TT; ++t) sum += __expf(L[(long long)t * HW] - mx);
    float inv = 1.f / sum;
    unsigned short* O = Abf + ((long long)n * HW + s) * TP;
    for (int t = 0; t < TT; ++t) O[t] = f2bf(__expf(L[(long long)t * HW] - mx) * inv);
    for (int t = TT; t < TP; ++t) O[t] = 0;
}

// ---- per-pixel 512-channel L2 norm: fea [n][hw][512] f32 -> fean bf16
__global__ __launch_bounds__(256)
void k_norm(const float* __restrict__ fea, unsigned short* __restrict__ fean) {
    long long idx = (long long)blockIdx.y * HW + blockIdx.x;
    const float* F = fea + idx * 512;
    unsigned short* O = fean + idx * 512;
    int t = threadIdx.x;
    float a = F[t], b = F[t + 256];
    __shared__ float red[256];
    red[t] = a * a + b * b;
    __syncthreads();
    for (int st = 128; st > 0; st >>= 1) { if (t < st) red[t] += red[t + st]; __syncthreads(); }
    float inv = 1.f / fmaxf(sqrtf(red[0]), 1e-12f);
    O[t] = f2bf(a * inv);
    O[t + 256] = f2bf(b * inv);
}

// ---- repack Wf [o][512][3][3] f32 -> Wfr bf16 [o][(ky*3+kx)*512 + i]
__global__ __launch_bounds__(256)
void k_repack(const float* __restrict__ Wf, unsigned short* __restrict__ Wfr) {
    int o = blockIdx.x;
    for (int idx = threadIdx.x; idx < KCONV; idx += 256) {
        int kq = idx >> 9, i = idx & 511;
        int ky = kq / 3, kx = kq % 3;
        Wfr[(long long)o * KCONV + idx] =
            f2bf(Wf[(((long long)o * 512 + i) * 3 + ky) * 3 + kx]);
    }
}

// ---- implicit-GEMM 3x3 valid conv + relu: fean [n][48*48][512] bf16 -> out [n][256][46*46] f32
__global__ __launch_bounds__(256)
void k_conv(const unsigned short* __restrict__ Wfr,
            const unsigned short* __restrict__ fean,
            float* __restrict__ out) {
    __shared__ __align__(16) unsigned short As[128 * 32];
    __shared__ __align__(16) unsigned short Bs[64 * 32];
    int m0 = blockIdx.x * 128, n0 = blockIdx.y * 64, n = blockIdx.z;
    const unsigned short* feaN = fean + (long long)n * HW * 512;
    int tid = threadIdx.x, lane = tid & 31, w = tid >> 5;
    int wm = (w & 1) * 64, wn = (w >> 1) * 16;
    v8f acc[4];
    for (int i = 0; i < 4; ++i) acc[i] = (v8f){0.f,0.f,0.f,0.f,0.f,0.f,0.f,0.f};
    for (int k0 = 0; k0 < KCONV; k0 += 32) {
        __syncthreads();
        {   // A: repacked weights, contiguous 16B chunks
            int row = tid >> 1, kb = (tid & 1) * 16;
            const unsigned short* g = Wfr + (long long)(m0 + row) * KCONV + k0 + kb;
#if USE_ASYNC_LDS
            async_cp16(g, &As[row * 32 + kb]);
            async_cp16(g + 8, &As[row * 32 + kb + 8]);
#else
#pragma unroll
            for (int j = 0; j < 16; ++j) As[row * 32 + kb + j] = g[j];
            if (k0 + 32 < KCONV) __builtin_prefetch((const void*)(g + 32), 0, 0);
#endif
        }
        {   // B: 32 contiguous channels of shifted pixel (k-chunk never crosses (ky,kx))
            int col = tid >> 2, kb = (tid & 3) * 8;
            int p = n0 + col;
            int pc = p < OP ? p : OP - 1;      // clamp; out-of-range cols are discarded at store
            int kq = k0 >> 9, i0 = (k0 & 511) + kb;
            int ky = kq / 3, kx = kq % 3;
            int y = pc / OHW + ky, x = pc % OHW + kx;
            const unsigned short* g = feaN + (long long)(y * HH + x) * 512 + i0;
#if USE_ASYNC_LDS
            async_cp16(g, &Bs[col * 32 + kb]);
#else
            unsigned short tmp[8];
#pragma unroll
            for (int j = 0; j < 8; ++j) tmp[j] = g[j];
            uint4 vv;
            vv.x = (unsigned)tmp[0] | ((unsigned)tmp[1] << 16);
            vv.y = (unsigned)tmp[2] | ((unsigned)tmp[3] << 16);
            vv.z = (unsigned)tmp[4] | ((unsigned)tmp[5] << 16);
            vv.w = (unsigned)tmp[6] | ((unsigned)tmp[7] << 16);
            *(uint4*)(&Bs[col * 32 + kb]) = vv;
#endif
        }
#if USE_ASYNC_LDS
        __builtin_amdgcn_s_wait_asynccnt(0);
#endif
        __syncthreads();
        v16bf b = frag_b(Bs, wn, lane);
#pragma unroll
        for (int mi = 0; mi < 4; ++mi) {
            v16bf a = frag_a(As, wm + mi * 16, lane);
            acc[mi] = __builtin_amdgcn_wmma_f32_16x16x32_bf16(
                false, a, false, b, (short)0, acc[mi], false, false);
        }
    }
    int p = n0 + wn + (lane & 15);
    if (p < OP) {
        int kh = (lane >> 4) * 8;
        float* o = out + (long long)n * CCH * OP;
#pragma unroll
        for (int mi = 0; mi < 4; ++mi)
#pragma unroll
            for (int i = 0; i < 8; ++i) {
                int row = m0 + wm + mi * 16 + i + kh;
                float v = acc[mi][i];
                o[(long long)row * OP + p] = v > 0.f ? v : 0.f;
            }
    }
}

extern "C" void kernel_launch(void* const* d_in, const int* in_sizes, int n_in,
                              void* d_out, int out_size, void* d_ws, size_t ws_size,
                              hipStream_t stream) {
    const float* x1  = (const float*)d_in[0];
    const float* x2  = (const float*)d_in[1];
    const float* Wm1 = (const float*)d_in[2];
    const float* Wm2 = (const float*)d_in[3];
    const float* Wm3 = (const float*)d_in[4];
    const float* Wl1 = (const float*)d_in[5];
    const float* Wl2 = (const float*)d_in[6];
    const float* Wl3 = (const float*)d_in[7];
    const float* Wf  = (const float*)d_in[8];
    float* out = (float*)d_out;

    char* ws = (char*)d_ws;
    size_t off = 0;
    auto alloc = [&](size_t bytes) -> void* {
        void* p = ws + off;
        off += (bytes + 255) & ~(size_t)255;
        return p;
    };
    float* pooled1 = (float*)alloc((size_t)NB * CCH * TP * 4);
    float* pooled2 = (float*)alloc((size_t)NB * CCH * TP * 4);
    unsigned short* SAM  = (unsigned short*)alloc((size_t)NB * TP * CCH * 2);
    unsigned short* SM3  = (unsigned short*)alloc((size_t)NB * TP * CCH * 2);
    unsigned short* SAL  = (unsigned short*)alloc((size_t)NB * TP * CCH * 2);
    unsigned short* SL3  = (unsigned short*)alloc((size_t)NB * TP * CCH * 2);
    unsigned short* M2bf = (unsigned short*)alloc((size_t)NB * CCH * HW * 2);
    unsigned short* L2bf = (unsigned short*)alloc((size_t)NB * CCH * HW * 2);
    float* logits = (float*)alloc((size_t)NB * TP * HW * 4);
    unsigned short* Abf  = (unsigned short*)alloc((size_t)NB * HW * TP * 2);
    float* fea = (float*)alloc((size_t)NB * HW * 512 * 4);
    unsigned short* fean = (unsigned short*)alloc((size_t)NB * HW * 512 * 2);
    unsigned short* Wfr  = (unsigned short*)alloc((size_t)CCH * KCONV * 2);

    // 1) pyramid pooling of inputs (exploits linearity of pool + 1x1 conv)
    k_pool<<<dim3(CCH, NB), 128, 0, stream>>>(x1, pooled1);
    k_pool<<<dim3(CCH, NB), 128, 0, stream>>>(x2, pooled2);
    // 2) small projections + relu + token-L2 norm -> SAM/SM3/SAL/SL3
    k_sproj<<<dim3(CCH, 4, NB), 128, 0, stream>>>(Wm1, Wm3, Wl1, Wl3,
                                                  pooled1, pooled2,
                                                  SAM, SM3, SAL, SL3);
    // 3) full-res projections M2 = Wm2*x1, L2 = Wl2*x2 (f32 in -> bf16 out)
    k_gemm<<<dim3(2, 36, NB), 256, 0, stream>>>(
        Wm2, 0LL, CCH, 1, x1, (long long)CCH * HW, HW, 1,
        nullptr, 0LL, nullptr, 0LL, 0,
        M2bf, (long long)CCH * HW, HW, 1, CCH);
    k_gemm<<<dim3(2, 36, NB), 256, 0, stream>>>(
        Wl2, 0LL, CCH, 1, x2, (long long)CCH * HW, HW, 1,
        nullptr, 0LL, nullptr, 0LL, 0,
        L2bf, (long long)CCH * HW, HW, 1, CCH);
    // 4) logits = SAM*M2 + SAL*L2 (dual-pair WMMA)
    k_gemm<<<dim3(1, 36, NB), 256, 0, stream>>>(
        SAM, (long long)TP * CCH, CCH, 0, M2bf, (long long)CCH * HW, HW, 0,
        SAL, (long long)TP * CCH, L2bf, (long long)CCH * HW, 1,
        logits, (long long)TP * HW, HW, 0, CCH);
    // 5) softmax over tokens -> bf16 attention, K-padded to 128
    k_softmax<<<dim3(HW / 128, NB), 128, 0, stream>>>(logits, Abf);
    // 6) attention outputs into channel-last [n][hw][512] buffer
    k_gemm<<<dim3(18, 4, NB), 256, 0, stream>>>(
        Abf, (long long)HW * TP, TP, 0, SM3, (long long)TP * CCH, CCH, 0,
        nullptr, 0LL, nullptr, 0LL, 0,
        fea, (long long)HW * 512, 512, 0, TP);
    k_gemm<<<dim3(18, 4, NB), 256, 0, stream>>>(
        Abf, (long long)HW * TP, TP, 0, SL3, (long long)TP * CCH, CCH, 0,
        nullptr, 0LL, nullptr, 0LL, 0,
        fea + 256, (long long)HW * 512, 512, 0, TP);
    // 7) per-pixel channel L2 norm -> bf16 NHWC
    k_norm<<<dim3(HW, NB), 256, 0, stream>>>(fea, fean);
    // 8) repack conv weights; implicit-GEMM 3x3 conv + relu -> out
    k_repack<<<dim3(CCH), 256, 0, stream>>>(Wf, Wfr);
    k_conv<<<dim3(2, 34, NB), 256, 0, stream>>>(Wfr, fean, out);
}
